// Mixing_48515950576232
// MI455X (gfx1250) — matrix operand order repaired and checked by reading, not compiled, not run on hardware
//
#include <hip/hip_runtime.h>
#include <hip/hip_bf16.h>

typedef __attribute__((ext_vector_type(16))) __bf16 v16bf;
typedef __attribute__((ext_vector_type(8)))  float  v8f;

#define Bq 8
#define Nn 4096
#define Dd 1024
#define BM 128
#define BN 128
#define BK 32
#define LDT 40   // LDS row stride in ushorts: 80 B, 16-B aligned, odd bank step

union Frag {
  v16bf v;
  uint4 q[2];
};

__device__ __forceinline__ unsigned short f32_to_bf16(float f) {
  unsigned int u = __float_as_uint(f);
  unsigned int r = u + 0x7FFFu + ((u >> 16) & 1u);   // round-to-nearest-even
  return (unsigned short)(r >> 16);
}

// CDNA5 async memory->LDS copy (16 B per lane), tracked with ASYNCcnt.
// LDS address operand is the wave-relative byte offset (low 32 bits of the
// generic pointer per the flat-aperture rule).
__device__ __forceinline__ void async_b128(unsigned lds_off, const void* gptr) {
  asm volatile("global_load_async_to_lds_b128 %0, %1, off"
               :: "v"(lds_off), "v"(gptr)
               : "memory");
}
__device__ __forceinline__ void wait_async0() {
  asm volatile("s_wait_asynccnt 0x0" ::: "memory");
}

// ---------------------------------------------------------------------------
// Kernel 1: materialize lower-triangular Toeplitz matrix T[n][m] = w[n-m] (bf16)
// One block per row n; each thread writes 16 contiguous elements (2 x b128).
// ---------------------------------------------------------------------------
__global__ __launch_bounds__(256) void toeplitz_bf16(const float* __restrict__ w,
                                                     unsigned short* __restrict__ T) {
  const int n  = blockIdx.x;
  const int m0 = threadIdx.x * 16;
  unsigned short vals[16];
#pragma unroll
  for (int i = 0; i < 16; ++i) {
    const int m = m0 + i;
    vals[i] = (m <= n) ? f32_to_bf16(w[n - m]) : (unsigned short)0;
  }
  uint4* dst = (uint4*)&T[(size_t)n * Nn + m0];
  dst[0] = ((const uint4*)vals)[0];
  dst[1] = ((const uint4*)vals)[1];
}

// ---------------------------------------------------------------------------
// Kernel 2: Xt[b][d][n] = bf16(x[b][n][d])  (tiled 32x32 transpose via LDS)
// ---------------------------------------------------------------------------
__global__ __launch_bounds__(256) void transpose_bf16(const float* __restrict__ x,
                                                      unsigned short* __restrict__ Xt) {
  __shared__ unsigned short tile[32][33];
  const int n0 = blockIdx.x * 32, d0 = blockIdx.y * 32, bb = blockIdx.z;
  const float* xb = x + (size_t)bb * Nn * Dd;
  unsigned short* Xtb = Xt + (size_t)bb * Dd * Nn;
  const int j = threadIdx.x & 31, i0 = threadIdx.x >> 5;
#pragma unroll
  for (int p = 0; p < 4; ++p) {
    const int i = i0 + p * 8;
    tile[i][j] = f32_to_bf16(xb[(size_t)(n0 + i) * Dd + d0 + j]);
  }
  __syncthreads();
  const int ii = threadIdx.x & 31, jj0 = threadIdx.x >> 5;
#pragma unroll
  for (int p = 0; p < 4; ++p) {
    const int jj = jj0 + p * 8;
    Xtb[(size_t)(d0 + jj) * Nn + n0 + ii] = tile[ii][jj];
  }
}

// ---------------------------------------------------------------------------
// Kernel 3: batched triangular GEMM  Y[b] = T @ X[b]  via v_wmma_f32_16x16x32_bf16
// 128x128 tile / 256 threads (8 waves, 2x4 wave grid, 64x32 per wave).
// Double-buffered LDS fed by global_load_async_to_lds_b128 (ASYNCcnt-tracked),
// triangular K-loop bound (nk = n0/32 + 4).
// ---------------------------------------------------------------------------
__global__ __launch_bounds__(256) void toeplitz_gemm_bf16(
    const unsigned short* __restrict__ T,
    const unsigned short* __restrict__ XtAll,
    float* __restrict__ Yall) {
  __shared__ unsigned short As[2][BM * LDT];
  __shared__ unsigned short Bs[2][BN * LDT];

  const int tid = threadIdx.x;
  const int n0 = blockIdx.x * BM;
  const int d0 = blockIdx.y * BN;
  const int bb = blockIdx.z;

  const unsigned short* Xt = XtAll + (size_t)bb * Dd * Nn;
  float* Y = Yall + (size_t)bb * Nn * Dd;

  // tile copy: 128 rows x 32 k (64 B/row) per operand; 4 async b128 per thread
  const int r0  = tid >> 2;
  const int off = (tid & 3) * 8;

  auto tile_copy = [&](int kb, int buf) {
    const size_t ka = (size_t)kb * BK + off;
    async_b128((unsigned)(size_t)&As[buf][(r0)      * LDT + off],
               &T[(size_t)(n0 + r0) * Nn + ka]);
    async_b128((unsigned)(size_t)&As[buf][(r0 + 64) * LDT + off],
               &T[(size_t)(n0 + r0 + 64) * Nn + ka]);
    async_b128((unsigned)(size_t)&Bs[buf][(r0)      * LDT + off],
               &Xt[(size_t)(d0 + r0) * Nn + ka]);
    async_b128((unsigned)(size_t)&Bs[buf][(r0 + 64) * LDT + off],
               &Xt[(size_t)(d0 + r0 + 64) * Nn + ka]);
  };

  const int lane = tid & 31, wid = tid >> 5;
  const int wm = (wid >> 2) * 64;   // wave row offset   (0,64)
  const int wn = (wid & 3) * 32;    // wave col offset   (0,32,64,96)
  const int ml = lane & 15, g = lane >> 4;

  v8f acc[4][2];
#pragma unroll
  for (int fm = 0; fm < 4; ++fm)
#pragma unroll
    for (int fn = 0; fn < 2; ++fn) acc[fm][fn] = (v8f)(0.0f);

  // triangular bound: T[n][k] == 0 for k > n  ->  only k-blocks with k0 < n0+BM
  const int nk = (n0 >> 5) + (BM / BK);

  tile_copy(0, 0);

  for (int kb = 0; kb < nk; ++kb) {
    const int cur = kb & 1;
    wait_async0();        // this wave's copies for tile kb have landed in LDS
    __syncthreads();      // all waves' copies visible; prev compute on buf^1 done
    if (kb + 1 < nk) tile_copy(kb + 1, cur ^ 1);   // overlap with compute below

    const unsigned short* A_ = As[cur];
    const unsigned short* B_ = Bs[cur];

    // B fragments: lane holds column d = ml, elements K = 16g..16g+15 (contiguous)
    Frag bfr[2];
#pragma unroll
    for (int fn = 0; fn < 2; ++fn) {
      const int d = wn + fn * 16 + ml;
      bfr[fn].q[0] = *(const uint4*)&B_[d * LDT + g * 16];
      bfr[fn].q[1] = *(const uint4*)&B_[d * LDT + g * 16 + 8];
    }
    // A fragments: lane holds row m = ml, elements K = 16g..16g+15 (contiguous)
#pragma unroll
    for (int fm = 0; fm < 4; ++fm) {
      Frag af;
      const int m = wm + fm * 16 + ml;
      af.q[0] = *(const uint4*)&A_[m * LDT + g * 16];
      af.q[1] = *(const uint4*)&A_[m * LDT + g * 16 + 8];
#pragma unroll
      for (int fn = 0; fn < 2; ++fn)
        acc[fm][fn] = __builtin_amdgcn_wmma_f32_16x16x32_bf16(
            false, af.v, false, bfr[fn].v, (short)0, acc[fm][fn], false, false);
    }
  }

  // epilogue: C/D layout -> VGPR r: row = r + 8*(lane>>4); col = lane&15
  const int col = lane & 15, rg = (lane >> 4) * 8;
#pragma unroll
  for (int fm = 0; fm < 4; ++fm)
#pragma unroll
    for (int fn = 0; fn < 2; ++fn) {
      const int dcol = d0 + wn + fn * 16 + col;
#pragma unroll
      for (int r = 0; r < 8; ++r) {
        const int nrow = n0 + wm + fm * 16 + rg + r;
        Y[(size_t)nrow * Dd + dcol] = acc[fm][fn][r];
      }
    }
}

// ---------------------------------------------------------------------------
// Kernel 4: out = x + LayerNorm_D(Y) * gamma + beta   (one block per (b,n) row)
// ---------------------------------------------------------------------------
__global__ __launch_bounds__(256) void ln_residual(const float* __restrict__ x,
                                                   const float* __restrict__ Y,
                                                   const float* __restrict__ gamma,
                                                   const float* __restrict__ beta,
                                                   float* __restrict__ out) {
  __shared__ float s1[256], s2[256];
  const size_t row = blockIdx.x;
  const int t = threadIdx.x;

  const float4 v = *(const float4*)(Y + row * Dd + t * 4);
  s1[t] = v.x + v.y + v.z + v.w;
  s2[t] = v.x * v.x + v.y * v.y + v.z * v.z + v.w * v.w;
  __syncthreads();
#pragma unroll
  for (int o = 128; o > 0; o >>= 1) {
    if (t < o) { s1[t] += s1[t + o]; s2[t] += s2[t + o]; }
    __syncthreads();
  }
  const float mean = s1[0] * (1.0f / Dd);
  const float var  = s2[0] * (1.0f / Dd) - mean * mean;
  const float inv  = rsqrtf(var + 1e-5f);

  const float4 xv = *(const float4*)(x + row * Dd + t * 4);
  const float4 g4 = *(const float4*)(gamma + t * 4);
  const float4 b4 = *(const float4*)(beta + t * 4);
  float4 o4;
  o4.x = xv.x + (v.x - mean) * inv * g4.x + b4.x;
  o4.y = xv.y + (v.y - mean) * inv * g4.y + b4.y;
  o4.z = xv.z + (v.z - mean) * inv * g4.z + b4.z;
  o4.w = xv.w + (v.w - mean) * inv * g4.w + b4.w;
  *(float4*)(out + row * Dd + t * 4) = o4;
}

// ---------------------------------------------------------------------------
extern "C" void kernel_launch(void* const* d_in, const int* in_sizes, int n_in,
                              void* d_out, int out_size, void* d_ws, size_t ws_size,
                              hipStream_t stream) {
  const float* x     = (const float*)d_in[0];   // [8,4096,1024]
  const float* w     = (const float*)d_in[1];   // [4096]
  const float* gamma = (const float*)d_in[2];   // [1024]
  const float* beta  = (const float*)d_in[3];   // [1024]
  float* out = (float*)d_out;

  // workspace layout: T bf16 (32 MB) | Xt bf16 (64 MB) | Y fp32 (128 MB)
  unsigned short* T  = (unsigned short*)d_ws;
  unsigned short* Xt = T + (size_t)Nn * Nn;
  float* Y = (float*)(Xt + (size_t)Bq * Dd * Nn);

  toeplitz_bf16<<<dim3(Nn), dim3(256), 0, stream>>>(w, T);
  transpose_bf16<<<dim3(Nn / 32, Dd / 32, Bq), dim3(256), 0, stream>>>(x, Xt);
  toeplitz_gemm_bf16<<<dim3(Nn / BM, Dd / BN, Bq), dim3(256), 0, stream>>>(T, Xt, Y);
  ln_residual<<<dim3(Bq * Nn), dim3(256), 0, stream>>>(x, Y, gamma, beta, out);
}
